// CausalSelfAttention_52029233824209
// MI455X (gfx1250) — compile-verified, matrix operand order checked by reading, not run on hardware
//
#include <hip/hip_runtime.h>

// ---------------------------------------------------------------------------
// CDNA5 (gfx1250) causal self-attention: QKV GEMM -> flash attention -> out GEMM
// v_wmma_f32_16x16x32_bf16 everywhere; async global->LDS staging where legal.
// ---------------------------------------------------------------------------

typedef __attribute__((ext_vector_type(16))) __bf16 v16bf;
typedef __attribute__((ext_vector_type(8)))  __bf16 v8bf;
typedef __attribute__((ext_vector_type(8)))  float  v8f;
typedef __attribute__((ext_vector_type(4)))  int    v4i;

#define B_  2
#define T_  4096
#define C_  768
#define H_  12
#define D_  64

#if __has_builtin(__builtin_amdgcn_global_load_async_to_lds_b128) && \
    __has_builtin(__builtin_amdgcn_s_wait_asynccnt)
#define USE_ASYNC_LDS 1
#else
#define USE_ASYNC_LDS 0
#endif

#if USE_ASYNC_LDS
typedef __attribute__((address_space(1))) v4i* gv4p;   // global int4*
typedef __attribute__((address_space(3))) v4i* lv4p;   // LDS int4*
typedef __attribute__((address_space(1))) void* gvp;
typedef __attribute__((address_space(3))) void* lvp;
static __device__ inline void async_cp16(const __bf16* g, __bf16* l) {
  // 16-byte global -> LDS copy through the async engine (ASYNCcnt).
  __builtin_amdgcn_global_load_async_to_lds_b128((gv4p)(gvp)g, (lv4p)(lvp)l, 0, 0);
}
static __device__ inline void async_wait0() {
  __builtin_amdgcn_s_wait_asynccnt(0);
}
#else
static __device__ inline void async_cp16(const __bf16* g, __bf16* l) {
  *(v8bf*)l = *(const v8bf*)g;
}
static __device__ inline void async_wait0() {}
#endif

static __device__ inline __bf16 f2bf(float f) {
  unsigned u = __builtin_bit_cast(unsigned, f);
  u += 0x7FFFu + ((u >> 16) & 1u);               // round-to-nearest-even
  unsigned short h = (unsigned short)(u >> 16);
  return __builtin_bit_cast(__bf16, h);
}

static __device__ inline unsigned pack2(float a, float b) {
  unsigned lo = __builtin_bit_cast(unsigned short, f2bf(a));
  unsigned hi = __builtin_bit_cast(unsigned short, f2bf(b));
  return lo | (hi << 16);
}

// A-fragment (16x32 bf16, MxK) per CDNA5 ISA layout:
//   lane L: M = L&15 ; K chunks [8*(L>>4), +8) and [16+8*(L>>4), +8)
// Same layout serves B (32x16, KxN) when memory is stored transposed [n][k].
static __device__ inline v16bf load_frag(const __bf16* base, int stride, int lane) {
  int m  = lane & 15;
  int kh = (lane >> 4) << 3;
  const __bf16* p = base + m * stride;
  v8bf lo = *(const v8bf*)(p + kh);
  v8bf hi = *(const v8bf*)(p + 16 + kh);
  v16bf r;
#pragma unroll
  for (int i = 0; i < 8; ++i) { r[i] = lo[i]; r[8 + i] = hi[i]; }
  return r;
}

static __device__ inline v8f wmma_bf16(v16bf a, v16bf b, v8f c) {
  return __builtin_amdgcn_wmma_f32_16x16x32_bf16(false, a, false, b, (short)0, c,
                                                 false, false);
}

#define LDK 80   // padded k-stride for 64-wide K staging (160B, 16B-aligned rows)

// ---------------------------------------------------------------------------
// Kernel 1: qkv = x @ w_qkv ; scatter into Q (x 1/sqrt(D)), K, V as (B,H,T,D).
// grid (8192/64, 2304/64), block 128. K staged 64 deep -> 8 WMMA per barrier.
// ---------------------------------------------------------------------------
__global__ __launch_bounds__(128)
void qkv_proj_kernel(const float* __restrict__ x, const float* __restrict__ w,
                     __bf16* __restrict__ Qb, __bf16* __restrict__ Kb,
                     __bf16* __restrict__ Vb) {
  __shared__ __align__(16) __bf16 As[64][LDK];   // [m][k]
  __shared__ __align__(16) __bf16 Bs[64][LDK];   // [n][k] (transposed)

  const int tid = threadIdx.x, lane = tid & 31, wid = tid >> 5;
  const int m0 = blockIdx.x * 64, n0 = blockIdx.y * 64;
  const int wm = (wid >> 1) * 32, wn = (wid & 1) * 32;

  v8f acc[2][2] = {};

  for (int kk = 0; kk < C_; kk += 64) {
    // stage A 64x64 (fp32 -> packed bf16, dword LDS stores)
    {
      int row = tid >> 1, cb = (tid & 1) * 32;
      const float* src = x + (size_t)(m0 + row) * C_ + kk + cb;
      unsigned* dst = (unsigned*)&As[row][cb];
#pragma unroll
      for (int i = 0; i < 16; ++i) dst[i] = pack2(src[2 * i], src[2 * i + 1]);
    }
    // stage B 64x64 transposed: thread owns a k-pair x 16 n values
    {
      int kp = (tid & 31) * 2, nb = (tid >> 5) * 16;
      const float* s0 = w + (size_t)(kk + kp) * (3 * C_) + n0 + nb;
      const float* s1 = s0 + 3 * C_;
#pragma unroll
      for (int i = 0; i < 16; ++i)
        *(unsigned*)&Bs[nb + i][kp] = pack2(s0[i], s1[i]);
    }
    __syncthreads();

#pragma unroll
    for (int ks = 0; ks < 2; ++ks) {
      v16bf a0 = load_frag(&As[wm][ks * 32],      LDK, lane);
      v16bf a1 = load_frag(&As[wm + 16][ks * 32], LDK, lane);
      v16bf b0 = load_frag(&Bs[wn][ks * 32],      LDK, lane);
      v16bf b1 = load_frag(&Bs[wn + 16][ks * 32], LDK, lane);
      acc[0][0] = wmma_bf16(a0, b0, acc[0][0]);
      acc[0][1] = wmma_bf16(a0, b1, acc[0][1]);
      acc[1][0] = wmma_bf16(a1, b0, acc[1][0]);
      acc[1][1] = wmma_bf16(a1, b1, acc[1][1]);
    }
    __syncthreads();
  }

  // epilogue: n-dependent work hoisted out of the row loop
  const int half = lane >> 4, nn = lane & 15;
  const int bidx  = m0 >> 12;            // tiles never straddle batch boundary
  const int tbase = m0 & (T_ - 1);
#pragma unroll
  for (int sn = 0; sn < 2; ++sn) {
    int n = n0 + wn + sn * 16 + nn;
    int which = n / C_, c = n % C_;
    int h = c >> 6, d = c & 63;
    __bf16* dst = (which == 0) ? Qb : (which == 1) ? Kb : Vb;
    float scale = (which == 0) ? 0.125f : 1.0f;   // 1/sqrt(64) folded into Q
    size_t base = (((size_t)bidx * H_ + h) * T_) * D_ + d;
#pragma unroll
    for (int sm = 0; sm < 2; ++sm)
#pragma unroll
      for (int r = 0; r < 8; ++r) {
        int t = tbase + wm + sm * 16 + r + 8 * half;
        dst[base + (size_t)t * D_] = f2bf(acc[sm][sn][r] * scale);
      }
  }
}

// ---------------------------------------------------------------------------
// Kernel 2: flash attention. grid (T/64, B*H), block 128 (4 waves x 16 rows).
// K tile staged via async global->LDS; V staged transposed with packed stores.
// ---------------------------------------------------------------------------
__global__ __launch_bounds__(128)
void flash_kernel(const __bf16* __restrict__ Qb, const __bf16* __restrict__ Kb,
                  const __bf16* __restrict__ Vb, __bf16* __restrict__ Y) {
  __shared__ __align__(16) __bf16 Ktile[64][64];      // [key][d]
  __shared__ __align__(16) __bf16 Vtile[64][64];      // [d][key] (transposed)
  __shared__ __align__(16) __bf16 Ptile[4][16][64];   // per-wave P re-layout

  const int tid = threadIdx.x, lane = tid & 31, wid = tid >> 5;
  const int qt = blockIdx.x;            // query tile 0..63
  const int bh = blockIdx.y;            // 0..23
  const int b = bh / H_, h = bh % H_;
  const int half = lane >> 4, nn = lane & 15;

  const __bf16* Qp = Qb + (size_t)bh * T_ * D_;
  const __bf16* Kp = Kb + (size_t)bh * T_ * D_;
  const __bf16* Vp = Vb + (size_t)bh * T_ * D_;

  const int q0 = qt * 64 + wid * 16;
  v16bf qf0 = load_frag(Qp + (size_t)q0 * D_,      D_, lane);  // d 0..31
  v16bf qf1 = load_frag(Qp + (size_t)q0 * D_ + 32, D_, lane);  // d 32..63

  v8f o[4] = {};
  float mi[8], li[8];
#pragma unroll
  for (int r = 0; r < 8; ++r) { mi[r] = -1e30f; li[r] = 0.f; }

  for (int j = 0; j <= qt; ++j) {
    __syncthreads();   // previous tile fully consumed
    {   // K tile: 4x16B async copies per thread (ASYNCcnt engine)
      int row = tid >> 1, cb = (tid & 1) * 32;
      const __bf16* ksrc = Kp + ((size_t)(j * 64 + row)) * D_ + cb;
      __bf16* kdst = &Ktile[row][cb];
#pragma unroll
      for (int i = 0; i < 4; ++i) async_cp16(ksrc + 8 * i, kdst + 8 * i);
    }
    {   // V tile transposed: thread owns a key-pair x 16 d; packed b32 stores
      int kp = (tid & 31) * 2, db = (tid >> 5) * 16;
      const __bf16* v0 = Vp + ((size_t)(j * 64 + kp)) * D_ + db;
      const __bf16* v1 = v0 + D_;
#pragma unroll
      for (int i = 0; i < 16; ++i) {
        unsigned lo = __builtin_bit_cast(unsigned short, v0[i]);
        unsigned hi = __builtin_bit_cast(unsigned short, v1[i]);
        *(unsigned*)&Vtile[db + i][kp] = lo | (hi << 16);
      }
    }
    if (j < qt) {   // pull next tiles toward L2
      __builtin_prefetch(Kp + (size_t)(j + 1) * 64 * D_, 0, 1);
      __builtin_prefetch(Vp + (size_t)(j + 1) * 64 * D_, 0, 1);
    }
    async_wait0();
    __syncthreads();

    // S = Q @ K^T  (16 rows x 64 keys per wave)
    v8f s[4] = {};
#pragma unroll
    for (int ks = 0; ks < 2; ++ks) {
      v16bf aq = ks ? qf1 : qf0;
#pragma unroll
      for (int fn = 0; fn < 4; ++fn) {
        v16bf bk = load_frag(&Ktile[fn * 16][ks * 32], 64, lane);
        s[fn] = wmma_bf16(aq, bk, s[fn]);
      }
    }

    if (j == qt) {  // causal mask inside the diagonal tile
#pragma unroll
      for (int fn = 0; fn < 4; ++fn)
#pragma unroll
        for (int r = 0; r < 8; ++r) {
          int kg = fn * 16 + nn;
          int qg = wid * 16 + r + 8 * half;
          if (kg > qg) s[fn][r] = -1e30f;
        }
    }

    // online softmax: row stats at vgpr r for rows r + 8*half
#pragma unroll
    for (int r = 0; r < 8; ++r) {
      float v = fmaxf(fmaxf(s[0][r], s[1][r]), fmaxf(s[2][r], s[3][r]));
#pragma unroll
      for (int off = 1; off < 16; off <<= 1)
        v = fmaxf(v, __shfl_xor(v, off, 32));
      float mnew  = fmaxf(mi[r], v);
      float alpha = __expf(mi[r] - mnew);
      mi[r] = mnew;
      li[r] *= alpha;
#pragma unroll
      for (int fd = 0; fd < 4; ++fd) o[fd][r] = o[fd][r] * alpha;
      float rs = 0.f;
#pragma unroll
      for (int fn = 0; fn < 4; ++fn) {
        float p = __expf(s[fn][r] - mnew);
        s[fn][r] = p;
        rs += p;
      }
#pragma unroll
      for (int off = 1; off < 16; off <<= 1) rs += __shfl_xor(rs, off, 32);
      li[r] += rs;
    }

    // re-layout P through per-wave LDS (same-wave DS ops are in-order)
#pragma unroll
    for (int fn = 0; fn < 4; ++fn)
#pragma unroll
      for (int r = 0; r < 8; ++r)
        Ptile[wid][r + 8 * half][fn * 16 + nn] = f2bf(s[fn][r]);

    // O += P @ V
#pragma unroll
    for (int ks = 0; ks < 2; ++ks) {
      v16bf ap = load_frag(&Ptile[wid][0][ks * 32], 64, lane);
#pragma unroll
      for (int fd = 0; fd < 4; ++fd) {
        v16bf bv = load_frag(&Vtile[fd * 16][ks * 32], 64, lane);
        o[fd] = wmma_bf16(ap, bv, o[fd]);
      }
    }
  }

  // normalize and store Y as (B, T, C) bf16
#pragma unroll
  for (int fd = 0; fd < 4; ++fd)
#pragma unroll
    for (int r = 0; r < 8; ++r) {
      float val = o[fd][r] / li[r];
      int q = qt * 64 + wid * 16 + r + 8 * half;
      int d = fd * 16 + nn;
      Y[((size_t)b * T_ + q) * C_ + h * D_ + d] = f2bf(val);
    }
}

// ---------------------------------------------------------------------------
// Kernel 3: out = Y @ w_out (bf16 x bf16 -> fp32). grid (8192/64, 768/64).
// A tile staged via async global->LDS (already bf16, no conversion needed).
// ---------------------------------------------------------------------------
__global__ __launch_bounds__(128)
void out_proj_kernel(const __bf16* __restrict__ Y, const float* __restrict__ w,
                     float* __restrict__ out) {
  __shared__ __align__(16) __bf16 As[64][LDK];
  __shared__ __align__(16) __bf16 Bs[64][LDK];

  const int tid = threadIdx.x, lane = tid & 31, wid = tid >> 5;
  const int m0 = blockIdx.x * 64, n0 = blockIdx.y * 64;
  const int wm = (wid >> 1) * 32, wn = (wid & 1) * 32;

  v8f acc[2][2] = {};

  for (int kk = 0; kk < C_; kk += 64) {
    {   // A tile: async bf16 copy, 4x16B per thread
      int row = tid >> 1, cb = (tid & 1) * 32;
      const __bf16* asrc = Y + (size_t)(m0 + row) * C_ + kk + cb;
      __bf16* adst = &As[row][cb];
#pragma unroll
      for (int i = 0; i < 4; ++i) async_cp16(asrc + 8 * i, adst + 8 * i);
    }
    {   // B tile transposed, packed dword stores
      int kp = (tid & 31) * 2, nb = (tid >> 5) * 16;
      const float* s0 = w + (size_t)(kk + kp) * C_ + n0 + nb;
      const float* s1 = s0 + C_;
#pragma unroll
      for (int i = 0; i < 16; ++i)
        *(unsigned*)&Bs[nb + i][kp] = pack2(s0[i], s1[i]);
    }
    async_wait0();
    __syncthreads();

#pragma unroll
    for (int ks = 0; ks < 2; ++ks) {
      v16bf a0 = load_frag(&As[wm][ks * 32],      LDK, lane);
      v16bf a1 = load_frag(&As[wm + 16][ks * 32], LDK, lane);
      v16bf b0 = load_frag(&Bs[wn][ks * 32],      LDK, lane);
      v16bf b1 = load_frag(&Bs[wn + 16][ks * 32], LDK, lane);
      acc[0][0] = wmma_bf16(a0, b0, acc[0][0]);
      acc[0][1] = wmma_bf16(a0, b1, acc[0][1]);
      acc[1][0] = wmma_bf16(a1, b0, acc[1][0]);
      acc[1][1] = wmma_bf16(a1, b1, acc[1][1]);
    }
    __syncthreads();
  }

  const int half = lane >> 4, nn = lane & 15;
#pragma unroll
  for (int sm = 0; sm < 2; ++sm)
#pragma unroll
    for (int sn = 0; sn < 2; ++sn)
#pragma unroll
      for (int r = 0; r < 8; ++r) {
        int m = m0 + wm + sm * 16 + r + 8 * half;
        int n = n0 + wn + sn * 16 + nn;
        out[(size_t)m * C_ + n] = acc[sm][sn][r];
      }
}

// ---------------------------------------------------------------------------
extern "C" void kernel_launch(void* const* d_in, const int* in_sizes, int n_in,
                              void* d_out, int out_size, void* d_ws, size_t ws_size,
                              hipStream_t stream) {
  (void)in_sizes; (void)n_in; (void)out_size; (void)ws_size;
  const float* x     = (const float*)d_in[0];
  const float* w_qkv = (const float*)d_in[1];
  const float* w_out = (const float*)d_in[2];
  float* out = (float*)d_out;

  const size_t NE = (size_t)B_ * T_ * C_;     // 6,291,456 elements per buffer
  __bf16* Qb = (__bf16*)d_ws;
  __bf16* Kb = Qb + NE;
  __bf16* Vb = Kb + NE;
  __bf16* Yb = Vb + NE;                        // total 4*NE*2 = ~50.3 MB

  dim3 g1((B_ * T_) / 64, (3 * C_) / 64);      // 128 x 36
  qkv_proj_kernel<<<g1, 128, 0, stream>>>(x, w_qkv, Qb, Kb, Vb);

  dim3 g2(T_ / 64, B_ * H_);                   // 64 x 24
  flash_kernel<<<g2, 128, 0, stream>>>(Qb, Kb, Vb, Yb);

  dim3 g3((B_ * T_) / 64, C_ / 64);            // 128 x 12
  out_proj_kernel<<<g3, 128, 0, stream>>>(Yb, w_out, out);
}
